// FigureTableInsertionModel_13314398617874
// MI455X (gfx1250) — compile-verified, matrix operand order checked by reading, not run on hardware
//
#include <hip/hip_runtime.h>
#include <hip/hip_bf16.h>

// ---------------- model dims ----------------
#define H_HEADS 8
#define D_HEAD  32
#define F_HID   256   // H*D
#define F_IN    256
#define F_OUT   128
#define N_PAR   100000
#define N_FIG   20000
#define N_TAB   20000
#define N_CIT   50000
#define N_EDGE  200000

typedef __attribute__((ext_vector_type(16))) _Float16 v16h;
typedef __attribute__((ext_vector_type(8)))  _Float16 v8h;
typedef __attribute__((ext_vector_type(8)))  float    v8f;

union Frag16 { v16h v; v8h h[2]; };

// =====================================================================
// WMMA GEMM, double-buffered LDS f16 tiles.
//   C[M,N] = A[M,K] @ B[K,N] + bias[N]
// Block: 256 threads (8 waves). Block tile 128x64, K-step 32.
// Pipeline: global->regs for tile t+1 issued before WMMAs of tile t;
// regs->LDS(buf (t+1)&1) after WMMAs; one barrier per K-step.
// Each wave computes a 16x64 strip: A fragment reused by 4 WMMAs.
// Fragment layout per CDNA5 ISA 7.12.2: lane(l%16)=row/col, hi=l/16,
// half-slot e -> K = (e%8) + 8*hi + 16*(e/8)  ==> two contiguous 16B runs.
// Requires N%64==0, K%32==0; M arbitrary (zero-padded in LDS).
// =====================================================================
#define BM 128
#define BN 64
#define BK 32
#define LDS_STRIDE 72   // halves per row; 144B: 16B-aligned, odd bank step

__global__ __launch_bounds__(256)
void wmma_gemm_bias(const float* __restrict__ A, const float* __restrict__ B,
                    const float* __restrict__ bias, float* __restrict__ C,
                    int M, int N, int K)
{
    __shared__ _Float16 lA[2][BM * LDS_STRIDE];   // [buf][row][k]
    __shared__ _Float16 lB[2][BN * LDS_STRIDE];   // [buf][col][k] (B transposed)

    const int tid  = threadIdx.x;
    const int wave = tid >> 5;
    const int lane = tid & 31;
    const int l16  = lane & 15;
    const int hi   = lane >> 4;

    const int bm = blockIdx.x * BM;
    const int bn = blockIdx.y * BN;

    // staging coords
    const int arow = tid >> 1;             // A: row 0..127
    const int ahalf = tid & 1;             // A: k offset 0/16
    const int bcol = tid & 63;             // B: col 0..63
    const int bkg  = tid >> 6;             // B: k group 0..3 (8 each)

    v8f acc[4] = {};
    v8h sa0, sa1, sb0;                      // register-staged tile data

    auto load_regs = [&](int kk) {
        const int gr = bm + arow;
        if (gr < M) {
            const float4* src = (const float4*)&A[(size_t)gr * K + kk + ahalf * 16];
            const float4 va = src[0], vb = src[1], vc = src[2], vd = src[3];
            sa0 = (v8h){ (_Float16)va.x, (_Float16)va.y, (_Float16)va.z, (_Float16)va.w,
                         (_Float16)vb.x, (_Float16)vb.y, (_Float16)vb.z, (_Float16)vb.w };
            sa1 = (v8h){ (_Float16)vc.x, (_Float16)vc.y, (_Float16)vc.z, (_Float16)vc.w,
                         (_Float16)vd.x, (_Float16)vd.y, (_Float16)vd.z, (_Float16)vd.w };
        } else {
            sa0 = (v8h){}; sa1 = (v8h){};
        }
        const float* bs = &B[(size_t)(kk + bkg * 8) * N + bn + bcol];
        sb0 = (v8h){ (_Float16)bs[0 * N], (_Float16)bs[1 * N],
                     (_Float16)bs[2 * N], (_Float16)bs[3 * N],
                     (_Float16)bs[4 * N], (_Float16)bs[5 * N],
                     (_Float16)bs[6 * N], (_Float16)bs[7 * N] };
    };
    auto store_lds = [&](int buf) {
        _Float16* da = &lA[buf][arow * LDS_STRIDE + ahalf * 16];
        *(v8h*)&da[0] = sa0;
        *(v8h*)&da[8] = sa1;
        *(v8h*)&lB[buf][bcol * LDS_STRIDE + bkg * 8] = sb0;
    };

    const int T = K / BK;
    load_regs(0);
    store_lds(0);

    for (int t = 0; t < T; ++t) {
        __syncthreads();                    // LDS buf (t&1) ready for all waves
        const int cur = t & 1;

        // prefetch tile t+2 into L2 (global_prefetch_b8)
        if (t + 2 < T) {
            if (bm + arow < M)
                __builtin_prefetch(&A[(size_t)(bm + arow) * K + (t + 2) * BK + ahalf * 16], 0, 0);
            __builtin_prefetch(&B[(size_t)((t + 2) * BK + bkg * 8) * N + bn + bcol], 0, 0);
        }
        // issue next tile's global loads before compute
        if (t + 1 < T) load_regs((t + 1) * BK);

        // fragments from LDS: A (2x b128) + 4x B (2x b128), then 4 WMMAs
        const int ar = wave * 16 + l16;
        Frag16 af;
        af.h[0] = *(const v8h*)&lA[cur][ar * LDS_STRIDE + 8 * hi];
        af.h[1] = *(const v8h*)&lA[cur][ar * LDS_STRIDE + 16 + 8 * hi];
        Frag16 bf[4];
#pragma unroll
        for (int s = 0; s < 4; ++s) {
            const int c = s * 16 + l16;
            bf[s].h[0] = *(const v8h*)&lB[cur][c * LDS_STRIDE + 8 * hi];
            bf[s].h[1] = *(const v8h*)&lB[cur][c * LDS_STRIDE + 16 + 8 * hi];
        }
#pragma unroll
        for (int s = 0; s < 4; ++s)
            acc[s] = __builtin_amdgcn_wmma_f32_16x16x32_f16(
                         false, af.v, false, bf[s].v, (short)0, acc[s], false, false);

        // write next tile into the other LDS buffer (no race: readers use buf cur)
        if (t + 1 < T) store_lds((t + 1) & 1);
    }

    // ---- store: C/D layout vgpr r -> M = r + 8*hi, lane%16 -> col ----
#pragma unroll
    for (int s = 0; s < 4; ++s) {
        const int col = bn + s * 16 + l16;
        const float bz = bias[col];
#pragma unroll
        for (int r = 0; r < 8; ++r) {
            const int orow = bm + wave * 16 + r + (hi << 3);
            if (orow < M) C[(size_t)orow * N + col] = acc[s][r] + bz;
        }
    }
}

// =====================================================================
// Utility fills
// =====================================================================
__global__ void fill_f32(float* p, float v, int n) {
    int i = blockIdx.x * blockDim.x + threadIdx.x;
    if (i < n) p[i] = v;
}
__global__ void fill_u32(unsigned* p, unsigned v, int n) {
    int i = blockIdx.x * blockDim.x + threadIdx.x;
    if (i < n) p[i] = v;
}

// =====================================================================
// Node attention scores: out[n,h] = sum_d h[n, h*32+d] * a[h,d]
// =====================================================================
__global__ void node_scores(const float* __restrict__ h, const float* __restrict__ aw,
                            float* __restrict__ out, int n_nodes)
{
    int i = blockIdx.x * blockDim.x + threadIdx.x;   // i = n*8 + head
    if (i >= n_nodes * H_HEADS) return;
    const int hh = i & (H_HEADS - 1);
    const float* hp = h + (size_t)(i >> 3) * F_HID + hh * D_HEAD;
    const float* ap = aw + hh * D_HEAD;
    float s = 0.f;
#pragma unroll
    for (int d = 0; d < D_HEAD; ++d) s += hp[d] * ap[d];
    out[i] = s;
}

// ordered-uint encoding so atomicMax(uint) == float max
__device__ __forceinline__ unsigned f2ord(float f) {
    unsigned u = __float_as_uint(f);
    return (u & 0x80000000u) ? ~u : (u | 0x80000000u);
}
__device__ __forceinline__ float ord2f(unsigned u) {
    return __uint_as_float((u & 0x80000000u) ? (u & 0x7FFFFFFFu) : ~u);
}
#define ORD_NEG_FLTMAX 0x00800000u   // f2ord(-FLT_MAX)

// =====================================================================
// Edge phase 1: alpha = leaky_relu(as[src]+ad[dst]); segment max via atomicMax
// =====================================================================
__global__ void edge_alpha(const float* __restrict__ as_, const float* __restrict__ ad_,
                           const int* __restrict__ si, const int* __restrict__ di,
                           float* __restrict__ alpha, unsigned* __restrict__ mx, int ne)
{
    int i = blockIdx.x * blockDim.x + threadIdx.x;   // i = e*8 + head
    if (i >= ne * H_HEADS) return;
    const int e = i >> 3, hh = i & 7;
    float a = as_[si[e] * H_HEADS + hh] + ad_[di[e] * H_HEADS + hh];
    a = a > 0.f ? a : 0.2f * a;                      // leaky_relu(0.2)
    alpha[i] = a;
    atomicMax(&mx[di[e] * H_HEADS + hh], f2ord(a));
}

// =====================================================================
// Edge phase 2: ex = exp(alpha - max); segment sum via atomicAdd
// =====================================================================
__global__ void edge_exp(const float* __restrict__ alpha, const int* __restrict__ di,
                         const unsigned* __restrict__ mx, float* __restrict__ ex,
                         float* __restrict__ den, int ne)
{
    int i = blockIdx.x * blockDim.x + threadIdx.x;
    if (i >= ne * H_HEADS) return;
    const int e = i >> 3, hh = i & 7;
    const float m = ord2f(mx[di[e] * H_HEADS + hh]);
    const float x = __expf(alpha[i] - m);
    ex[i] = x;
    atomicAdd(&den[di[e] * H_HEADS + hh], x);
}

// =====================================================================
// Edge phase 3: out[dst, f] += h_src[src, f] * (ex/den)[head(f)]
// one block (256 thr = 256 features) per edge
// =====================================================================
__global__ __launch_bounds__(256)
void edge_scatter(const float* __restrict__ h_src, const float* __restrict__ ex,
                  const float* __restrict__ den, const int* __restrict__ si,
                  const int* __restrict__ di, float* __restrict__ out, int ne)
{
    __shared__ float w[H_HEADS];
    __shared__ int sd[2];
    const int e = blockIdx.x;
    if (e >= ne) return;
    if (threadIdx.x == 0) { sd[0] = si[e]; sd[1] = di[e]; }
    __syncthreads();
    const int s = sd[0], d = sd[1];
    if (threadIdx.x < H_HEADS)
        w[threadIdx.x] = ex[e * H_HEADS + threadIdx.x] /
                         (den[d * H_HEADS + threadIdx.x] + 1e-16f);
    __syncthreads();
    const int f = threadIdx.x;
    atomicAdd(&out[(size_t)d * F_HID + f], h_src[(size_t)s * F_HID + f] * w[f >> 5]);
}

__global__ void relu_inplace(float* p, int n) {
    int i = blockIdx.x * blockDim.x + threadIdx.x;
    if (i < n) p[i] = fmaxf(p[i], 0.f);
}

// =====================================================================
// Semantic attention pieces
// =====================================================================
__global__ __launch_bounds__(256)
void tanh_colmean(const float* __restrict__ t, float* __restrict__ out, int n_rows)
{
    const int f = threadIdx.x;
    float s = 0.f;
    for (int n = blockIdx.x; n < n_rows; n += gridDim.x)
        s += tanhf(t[(size_t)n * F_HID + f]);
    atomicAdd(&out[f], s * (1.0f / (float)n_rows));
}

__global__ void semantic_attn(const float* __restrict__ kmean, const float* __restrict__ q,
                              float* __restrict__ attn)
{
    if (threadIdx.x == 0 && blockIdx.x == 0) {
        float sc[3]; float mx = -1e30f;
        for (int m = 0; m < 3; ++m) {
            float s = 0.f;
            for (int f = 0; f < F_HID; ++f) s += kmean[m * F_HID + f] * q[f];
            sc[m] = s; mx = fmaxf(mx, s);
        }
        float dn = 0.f;
        for (int m = 0; m < 3; ++m) { sc[m] = __expf(sc[m] - mx); dn += sc[m]; }
        for (int m = 0; m < 3; ++m) attn[m] = sc[m] / dn;
    }
}

__global__ void combine3(const float* __restrict__ s0, const float* __restrict__ s1,
                         const float* __restrict__ s2, const float* __restrict__ attn,
                         float* __restrict__ o, int n)
{
    int i = blockIdx.x * blockDim.x + threadIdx.x;
    if (i >= n) return;
    o[i] = attn[0] * s0[i] + attn[1] * s1[i] + attn[2] * s2[i];
}

// =====================================================================
// Classifier
// =====================================================================
__global__ void cls_target(const float* __restrict__ fig_emb, const float* __restrict__ cls_w,
                           const float* __restrict__ cls_b, const int* __restrict__ tgt,
                           float* __restrict__ tconst)
{
    if (threadIdx.x == 0 && blockIdx.x == 0) {
        const float* t = fig_emb + (size_t)tgt[0] * F_OUT;
        float s = cls_b[0];
        for (int f = 0; f < F_OUT; ++f) s += t[f] * cls_w[f];
        tconst[0] = s;
    }
}
__global__ void cls_scores(const float* __restrict__ par_emb, const float* __restrict__ cls_w,
                           const float* __restrict__ tconst, float* __restrict__ scores, int n)
{
    int i = blockIdx.x * blockDim.x + threadIdx.x;
    if (i >= n) return;
    float s = tconst[0];
    const float* p = par_emb + (size_t)i * F_OUT;
#pragma unroll 4
    for (int f = 0; f < F_OUT; ++f) s += p[f] * cls_w[F_OUT + f];
    scores[i] = s;
}

// =====================================================================
// Host side
// =====================================================================
extern "C" void kernel_launch(void* const* d_in, const int* in_sizes, int n_in,
                              void* d_out, int out_size, void* d_ws, size_t ws_size,
                              hipStream_t stream)
{
    // ---- inputs (setup_inputs order) ----
    const float* x_par = (const float*)d_in[0];
    const float* x_fig = (const float*)d_in[1];
    const float* x_tab = (const float*)d_in[2];
    const float* x_cit = (const float*)d_in[3];
    const float* Wp[4]  = {(const float*)d_in[4],  (const float*)d_in[6],
                           (const float*)d_in[8],  (const float*)d_in[10]};
    const float* bp[4]  = {(const float*)d_in[5],  (const float*)d_in[7],
                           (const float*)d_in[9],  (const float*)d_in[11]};
    // edge-type attention vectors: pf, fp, pt, tp, pc, cp
    const float* a_s[6] = {(const float*)d_in[12], (const float*)d_in[14],
                           (const float*)d_in[16], (const float*)d_in[18],
                           (const float*)d_in[20], (const float*)d_in[22]};
    const float* a_d[6] = {(const float*)d_in[13], (const float*)d_in[15],
                           (const float*)d_in[17], (const float*)d_in[19],
                           (const float*)d_in[21], (const float*)d_in[23]};
    const float* k_w  = (const float*)d_in[24];
    const float* k_b  = (const float*)d_in[25];
    const float* q    = (const float*)d_in[26];
    const float* Wh_par = (const float*)d_in[27];
    const float* bh_par = (const float*)d_in[28];
    const float* Wh_fig = (const float*)d_in[29];
    const float* bh_fig = (const float*)d_in[30];
    const float* cls_w  = (const float*)d_in[35];
    const float* cls_b  = (const float*)d_in[36];
    const int* e_src[6] = {(const int*)d_in[37], (const int*)d_in[39],
                           (const int*)d_in[41], (const int*)d_in[43],
                           (const int*)d_in[45], (const int*)d_in[47]};
    const int* e_dst[6] = {(const int*)d_in[38], (const int*)d_in[40],
                           (const int*)d_in[42], (const int*)d_in[44],
                           (const int*)d_in[46], (const int*)d_in[48]};
    const int* tgt_idx  = (const int*)d_in[49];

    // ---- workspace layout (float element offsets) ----
    float* ws = (float*)d_ws;
    const size_t HP = (size_t)N_PAR * F_HID, HF = (size_t)N_FIG * F_HID,
                 HT = (size_t)N_TAB * F_HID, HC = (size_t)N_CIT * F_HID;
    float* h_par   = ws + 0;
    float* h_fig   = h_par + HP;
    float* h_tab   = h_fig + HF;
    float* h_cit   = h_tab + HT;
    float* agg_fig = h_cit + HC;
    float* agg_tab = agg_fig + HF;
    float* agg_cit = agg_tab + HT;
    float* agg_p0  = agg_cit + HC;
    float* agg_p1  = agg_p0 + HP;
    float* agg_p2  = agg_p1 + HP;
    float* as_buf  = agg_p2 + HP;                       // 100000*8
    float* ad_buf  = as_buf + (size_t)N_PAR * H_HEADS;
    unsigned* mx_buf = (unsigned*)(ad_buf + (size_t)N_PAR * H_HEADS);
    float* den_buf = (float*)mx_buf + (size_t)N_PAR * H_HEADS;
    float* alpha   = den_buf + (size_t)N_PAR * H_HEADS;
    float* ex_buf  = alpha + (size_t)N_EDGE * H_HEADS;
    float* kmean   = ex_buf + (size_t)N_EDGE * H_HEADS; // 3*256
    float* attn    = kmean + 3 * F_HID;                 // 3
    float* tconst  = attn + 4;                          // 1
    // reuse: semantic temp -> h_par region; par_emb/fig_emb -> h_fig.. region
    float* sem_t   = h_par;
    float* par_emb = h_fig;                             // 100000*128
    float* fig_emb = par_emb + (size_t)N_PAR * F_OUT;   // 20000*128

    auto cdiv = [](long a, long b) { return (int)((a + b - 1) / b); };
    auto gemm = [&](const float* A, const float* B, const float* bias, float* C,
                    int M, int N, int K) {
        dim3 grid(cdiv(M, BM), N / BN);
        wmma_gemm_bias<<<grid, 256, 0, stream>>>(A, B, bias, C, M, N, K);
    };

    // ---- 1. zero aggregation targets ----
    {
        float* zs[6]  = {agg_fig, agg_tab, agg_cit, agg_p0, agg_p1, agg_p2};
        size_t  zn[6] = {HF, HT, HC, HP, HP, HP};
        for (int i = 0; i < 6; ++i)
            fill_f32<<<cdiv((long)zn[i], 256), 256, 0, stream>>>(zs[i], 0.f, (int)zn[i]);
        fill_f32<<<cdiv(3 * F_HID, 256), 256, 0, stream>>>(kmean, 0.f, 3 * F_HID);
    }

    // ---- 2. input projections via WMMA ----
    gemm(x_par, Wp[0], bp[0], h_par, N_PAR, F_HID, F_IN);
    gemm(x_fig, Wp[1], bp[1], h_fig, N_FIG, F_HID, F_IN);
    gemm(x_tab, Wp[2], bp[2], h_tab, N_TAB, F_HID, F_IN);
    gemm(x_cit, Wp[3], bp[3], h_cit, N_CIT, F_HID, F_IN);

    // ---- 3. six metapath aggregations ----
    // order matches a_s/a_d/e_src/e_dst: pf, fp, pt, tp, pc, cp
    const float* hsrc[6] = {h_par, h_fig, h_par, h_tab, h_par, h_cit};
    const float* hdst[6] = {h_fig, h_par, h_tab, h_par, h_cit, h_par};
    const int    Ns[6]   = {N_PAR, N_FIG, N_PAR, N_TAB, N_PAR, N_CIT};
    const int    Nd[6]   = {N_FIG, N_PAR, N_TAB, N_PAR, N_CIT, N_PAR};
    float*       aggb[6] = {agg_fig, agg_p0, agg_tab, agg_p1, agg_cit, agg_p2};

    const int egrid = cdiv((long)N_EDGE * H_HEADS, 256);
    for (int t = 0; t < 6; ++t) {
        node_scores<<<cdiv((long)Ns[t] * H_HEADS, 256), 256, 0, stream>>>(
            hsrc[t], a_s[t], as_buf, Ns[t]);
        node_scores<<<cdiv((long)Nd[t] * H_HEADS, 256), 256, 0, stream>>>(
            hdst[t], a_d[t], ad_buf, Nd[t]);
        fill_u32<<<cdiv((long)Nd[t] * H_HEADS, 256), 256, 0, stream>>>(
            mx_buf, ORD_NEG_FLTMAX, Nd[t] * H_HEADS);
        fill_f32<<<cdiv((long)Nd[t] * H_HEADS, 256), 256, 0, stream>>>(
            den_buf, 0.f, Nd[t] * H_HEADS);
        edge_alpha<<<egrid, 256, 0, stream>>>(as_buf, ad_buf, e_src[t], e_dst[t],
                                              alpha, mx_buf, N_EDGE);
        edge_exp<<<egrid, 256, 0, stream>>>(alpha, e_dst[t], mx_buf, ex_buf,
                                            den_buf, N_EDGE);
        edge_scatter<<<N_EDGE, 256, 0, stream>>>(hsrc[t], ex_buf, den_buf,
                                                 e_src[t], e_dst[t], aggb[t], N_EDGE);
        relu_inplace<<<cdiv((long)Nd[t] * F_HID, 256), 256, 0, stream>>>(
            aggb[t], Nd[t] * F_HID);
    }

    // ---- 4. semantic (metapath-level) attention for 'par' ----
    // (single-metapath groups are exactly identity under softmax of one logit)
    float* stp[3] = {agg_p0, agg_p1, agg_p2};
    for (int m = 0; m < 3; ++m) {
        gemm(stp[m], k_w, k_b, sem_t, N_PAR, F_HID, F_HID);    // WMMA
        tanh_colmean<<<256, 256, 0, stream>>>(sem_t, kmean + m * F_HID, N_PAR);
    }
    semantic_attn<<<1, 32, 0, stream>>>(kmean, q, attn);
    combine3<<<cdiv((long)HP, 256), 256, 0, stream>>>(agg_p0, agg_p1, agg_p2,
                                                      attn, agg_p0, (int)HP);

    // ---- 5. output heads via WMMA ----
    gemm(agg_p0, Wh_par, bh_par, par_emb, N_PAR, F_OUT, F_HID);
    gemm(agg_fig, Wh_fig, bh_fig, fig_emb, N_FIG, F_OUT, F_HID);

    // ---- 6. classifier ----
    cls_target<<<1, 32, 0, stream>>>(fig_emb, cls_w, cls_b, tgt_idx, tconst);
    cls_scores<<<cdiv(N_PAR, 256), 256, 0, stream>>>(par_emb, cls_w, tconst,
                                                     (float*)d_out, N_PAR);
}
